// Transformer_12283606467772
// MI455X (gfx1250) — matrix operand order repaired and checked by reading, not compiled, not run on hardware
//
#include <hip/hip_runtime.h>
#include <stdint.h>

// ---------------------------------------------------------------------------
// GTrXL transformer block for MI455X (gfx1250), wave32 + WMMA bf16.
// D=1024, NH=16, DH=64, B=16, QLEN=256, MEM=256, KLEN=512, SCALE=0.125
// ---------------------------------------------------------------------------

typedef __bf16 bf16_t;
typedef __attribute__((ext_vector_type(16))) __bf16 v16bf;
typedef __attribute__((ext_vector_type(8)))  __bf16 v8bf;
typedef __attribute__((ext_vector_type(8)))  float  v8f;

// ---- WMMA fragment helpers (k contiguous per lane half, same in A and B) ----------------------
// A (16x32 bf16): lane l: row = l&15; half h=l>>4: K = h*16 .. h*16+15 (contiguous)
__device__ __forceinline__ v16bf load_fragA(const bf16_t* rowp, int hl) {
  return *(const v16bf*)(rowp + hl * 16);
}
// B (32x16 bf16): lane l: col = l&15; half h=l>>4: K = h*16 .. h*16+15 (contiguous)
__device__ __forceinline__ v16bf load_fragB(const bf16_t* rowp, int hl) {
  return *(const v16bf*)(rowp + hl * 16);
}
__device__ __forceinline__ v8f wmma_bf(v16bf a, v16bf b, v8f c) {
  return __builtin_amdgcn_wmma_f32_16x16x32_bf16(false, a, false, b, (short)0, c,
                                                 false, false);
}
__device__ __forceinline__ v8f zero8() {
  v8f z = {0.f, 0.f, 0.f, 0.f, 0.f, 0.f, 0.f, 0.f};
  return z;
}

// ---------------------------------------------------------------------------
// Generic GEMM: C(MxN) = X(MxK) @ W(NxK)^T, bf16 in, f32 acc.
// One wave computes a 64x64 tile (4x4 WMMA tiles). Optional f32 accumulate
// and/or bf16 store.
// ---------------------------------------------------------------------------
__global__ __launch_bounds__(256) void gemm64x64(
    const bf16_t* __restrict__ X, int lda,
    const bf16_t* __restrict__ W, int ldb,
    float* __restrict__ C, bf16_t* __restrict__ Cb,
    int ldc, int M, int N, int K, int accumulate)
{
  const int lane = threadIdx.x & 31;
  const int wave = blockIdx.x * (blockDim.x >> 5) + (threadIdx.x >> 5);
  const int tilesN = N >> 6;
  const int tilesM = M >> 6;
  if (wave >= tilesM * tilesN) return;
  const int tm = (wave / tilesN) << 6;
  const int tn = (wave % tilesN) << 6;
  const int hl = lane >> 4, l16 = lane & 15;

  v8f acc[4][4];
#pragma unroll
  for (int r = 0; r < 4; ++r)
#pragma unroll
    for (int c = 0; c < 4; ++c) acc[r][c] = zero8();

  const bf16_t* aBase = X + (size_t)(tm + l16) * lda;
  const bf16_t* bBase = W + (size_t)(tn + l16) * ldb;

  for (int k0 = 0; k0 < K; k0 += 32) {
    v16bf a[4], b[4];
#pragma unroll
    for (int r = 0; r < 4; ++r) a[r] = load_fragA(aBase + (size_t)r * 16 * lda + k0, hl);
#pragma unroll
    for (int c = 0; c < 4; ++c) b[c] = load_fragB(bBase + (size_t)c * 16 * ldb + k0, hl);
    __builtin_prefetch(aBase + k0 + 32, 0, 1);   // global_prefetch_b8: next K tile
    __builtin_prefetch(bBase + k0 + 32, 0, 1);
#pragma unroll
    for (int r = 0; r < 4; ++r)
#pragma unroll
      for (int c = 0; c < 4; ++c) acc[r][c] = wmma_bf(a[r], b[c], acc[r][c]);
  }

#pragma unroll
  for (int r = 0; r < 4; ++r)
#pragma unroll
    for (int c = 0; c < 4; ++c) {
      const int col = tn + c * 16 + l16;
#pragma unroll
      for (int v = 0; v < 8; ++v) {
        const int row = tm + r * 16 + hl * 8 + v;
        const size_t idx = (size_t)row * ldc + col;
        float val = acc[r][c][v];
        if (C) {
          if (accumulate) val += C[idx];
          C[idx] = val;
        }
        if (Cb) Cb[idx] = (bf16_t)val;
      }
    }
}

// ---------------------------------------------------------------------------
// Per-head projection: rows of 64 through a 64x64 weight (X(Mx64) @ W(64x64)^T).
// mode 0: row-major bf16 out[row*64+col]
// mode 1: per-(b,n) transposed out for wv: [((b*16+n)*64+col)*512 + k]
// ---------------------------------------------------------------------------
__global__ __launch_bounds__(256) void proj16(
    const bf16_t* __restrict__ X, const bf16_t* __restrict__ W,
    bf16_t* __restrict__ Out, int mode, int Mrows)
{
  const int lane = threadIdx.x & 31;
  const int wave = blockIdx.x * (blockDim.x >> 5) + (threadIdx.x >> 5);
  const int tm = wave << 4;
  if (tm >= Mrows) return;
  const int hl = lane >> 4, l16 = lane & 15;

  v8f acc[4];
#pragma unroll
  for (int c = 0; c < 4; ++c) acc[c] = zero8();

  const bf16_t* aBase = X + (size_t)(tm + l16) * 64;
#pragma unroll
  for (int ks = 0; ks < 2; ++ks) {
    const int k0 = ks * 32;
    v16bf a = load_fragA(aBase + k0, hl);
#pragma unroll
    for (int c = 0; c < 4; ++c) {
      v16bf b = load_fragB(W + (size_t)(c * 16 + l16) * 64 + k0, hl);
      acc[c] = wmma_bf(a, b, acc[c]);
    }
  }
#pragma unroll
  for (int c = 0; c < 4; ++c) {
    const int col = c * 16 + l16;
#pragma unroll
    for (int v = 0; v < 8; ++v) {
      const int row = tm + hl * 8 + v;
      const float val = acc[c][v];
      if (mode == 0) {
        Out[(size_t)row * 64 + col] = (bf16_t)val;
      } else {  // transpose per (b,n): row = (b*512+k)*16+n
        const int b = row >> 13, k = (row >> 4) & 511, n = row & 15;
        Out[((size_t)((b * 16 + n) * 64 + col)) * 512 + k] = (bf16_t)val;
      }
    }
  }
}

// ---------------------------------------------------------------------------
// AC: per (b,n) batch, score[i,j] = sum_d qa[i,d]*wk[j,d]
// qa row i @ qa + i*16384 + bn*64 ; wk row j @ wk + j*16384 + bn*64
// ---------------------------------------------------------------------------
__global__ __launch_bounds__(256) void attn_ac(
    const bf16_t* __restrict__ qa, const bf16_t* __restrict__ wk,
    float* __restrict__ score)
{
  const int lane = threadIdx.x & 31;
  const int wave = blockIdx.x * (blockDim.x >> 5) + (threadIdx.x >> 5);
  const int bn = wave >> 7;            // 256 batches * 128 waves/batch
  const int rem = wave & 127;
  const int i0 = (rem >> 3) << 4;
  const int j0 = (rem & 7) << 6;
  const int hl = lane >> 4, l16 = lane & 15;

  v8f acc[4];
#pragma unroll
  for (int c = 0; c < 4; ++c) acc[c] = zero8();

  const bf16_t* aBase = qa + (size_t)(i0 + l16) * 16384 + bn * 64;
  const bf16_t* bBase = wk + (size_t)(j0 + l16) * 16384 + bn * 64;
#pragma unroll
  for (int ks = 0; ks < 2; ++ks) {
    const int k0 = ks * 32;
    v16bf a = load_fragA(aBase + k0, hl);
#pragma unroll
    for (int c = 0; c < 4; ++c) {
      v16bf b = load_fragB(bBase + (size_t)c * 16 * 16384 + k0, hl);
      acc[c] = wmma_bf(a, b, acc[c]);
    }
  }
  float* s = score + (size_t)bn * 131072;
#pragma unroll
  for (int c = 0; c < 4; ++c) {
    const int j = j0 + c * 16 + l16;
#pragma unroll
    for (int v = 0; v < 8; ++v) {
      const int i = i0 + hl * 8 + v;
      s[(size_t)i * 512 + j] = acc[c][v];
    }
  }
}

// ---------------------------------------------------------------------------
// BD with rel_shift folded into the store: raw (i,j) -> g = i*512 + (i+j-255),
// scatter-added into score (mapping is injective; pad slots receive nothing).
// rk depends only on n: row j @ rk + j*1024 + n*64
// ---------------------------------------------------------------------------
__global__ __launch_bounds__(256) void attn_bd(
    const bf16_t* __restrict__ qb, const bf16_t* __restrict__ rk,
    float* __restrict__ score)
{
  const int lane = threadIdx.x & 31;
  const int wave = blockIdx.x * (blockDim.x >> 5) + (threadIdx.x >> 5);
  const int bn = wave >> 7;
  const int rem = wave & 127;
  const int i0 = (rem >> 3) << 4;
  const int j0 = (rem & 7) << 6;
  const int n = bn & 15;
  const int hl = lane >> 4, l16 = lane & 15;

  v8f acc[4];
#pragma unroll
  for (int c = 0; c < 4; ++c) acc[c] = zero8();

  const bf16_t* aBase = qb + (size_t)(i0 + l16) * 16384 + bn * 64;
  const bf16_t* bBase = rk + (size_t)(j0 + l16) * 1024 + n * 64;
#pragma unroll
  for (int ks = 0; ks < 2; ++ks) {
    const int k0 = ks * 32;
    v16bf a = load_fragA(aBase + k0, hl);
#pragma unroll
    for (int c = 0; c < 4; ++c) {
      v16bf b = load_fragB(bBase + (size_t)c * 16 * 1024 + k0, hl);
      acc[c] = wmma_bf(a, b, acc[c]);
    }
  }
  float* s = score + (size_t)bn * 131072;
#pragma unroll
  for (int c = 0; c < 4; ++c) {
    const int j = j0 + c * 16 + l16;
#pragma unroll
    for (int v = 0; v < 8; ++v) {
      const int i = i0 + hl * 8 + v;
      const long g = (long)i * 512 + (i + j - 255);
      if (g >= 0 && g < 131072) s[g] += acc[c][v];
    }
  }
}

// ---------------------------------------------------------------------------
// vec: per (b,n): V(256x64) = P(256x512) @ wvT(64x512)^T ; store bf16 to
// vecb[i*16384 + bn*64 + d] (== ibnd flat == GRU token-major (4096,1024)).
// ---------------------------------------------------------------------------
__global__ __launch_bounds__(256) void attn_vec(
    const bf16_t* __restrict__ P, const bf16_t* __restrict__ wvT,
    bf16_t* __restrict__ vecb)
{
  const int lane = threadIdx.x & 31;
  const int wave = blockIdx.x * (blockDim.x >> 5) + (threadIdx.x >> 5);
  const int bn = wave >> 4;            // 256 batches * 16 i-tiles
  const int i0 = (wave & 15) << 4;
  const int hl = lane >> 4, l16 = lane & 15;

  v8f acc[4];
#pragma unroll
  for (int c = 0; c < 4; ++c) acc[c] = zero8();

  const bf16_t* aBase = P + (size_t)bn * 131072 + (size_t)(i0 + l16) * 512;
  const bf16_t* bBase = wvT + (size_t)bn * 32768 + (size_t)l16 * 512;
  for (int k0 = 0; k0 < 512; k0 += 32) {
    v16bf a = load_fragA(aBase + k0, hl);
    __builtin_prefetch(aBase + k0 + 32, 0, 1);
#pragma unroll
    for (int c = 0; c < 4; ++c) {
      v16bf b = load_fragB(bBase + (size_t)c * 16 * 512 + k0, hl);
      acc[c] = wmma_bf(a, b, acc[c]);
    }
  }
#pragma unroll
  for (int c = 0; c < 4; ++c) {
    const int d = c * 16 + l16;
#pragma unroll
    for (int v = 0; v < 8; ++v) {
      const int i = i0 + hl * 8 + v;
      vecb[(size_t)i * 16384 + bn * 64 + d] = (bf16_t)acc[c][v];
    }
  }
}

// ---------------------------------------------------------------------------
// LayerNorm over rows of 1024; dst row = (r/group)*ostride + ooff + (r%group)
// (used to build kv = concat(LN(value), LN(query)) directly).
// ---------------------------------------------------------------------------
__global__ __launch_bounds__(256) void ln_rows(
    const float* __restrict__ src, const float* __restrict__ g,
    const float* __restrict__ be, bf16_t* __restrict__ dst,
    int group, int ostride, int ooff)
{
  const int r = blockIdx.x;
  const int tid = threadIdx.x;
  const float* x = src + (size_t)r * 1024;
  float vals[4], s = 0.f, sq = 0.f;
#pragma unroll
  for (int e = 0; e < 4; ++e) {
    const float v = x[tid + e * 256];
    vals[e] = v; s += v; sq += v * v;
  }
  __shared__ float sh1[256], sh2[256];
  sh1[tid] = s; sh2[tid] = sq; __syncthreads();
  for (int st = 128; st > 0; st >>= 1) {
    if (tid < st) { sh1[tid] += sh1[tid + st]; sh2[tid] += sh2[tid + st]; }
    __syncthreads();
  }
  const float mean = sh1[0] * (1.f / 1024.f);
  const float var  = sh2[0] * (1.f / 1024.f) - mean * mean;
  const float rstd = rsqrtf(var + 1e-5f);
  const size_t orow = (size_t)(r / group) * ostride + ooff + (r % group);
  bf16_t* o = dst + orow * 1024;
#pragma unroll
  for (int e = 0; e < 4; ++e) {
    const int d = tid + e * 256;
    o[d] = (bf16_t)((vals[e] - mean) * rstd * g[d] + be[d]);
  }
}

// ---------------------------------------------------------------------------
// Masked scaled softmax over klen=512; prob f32 (i,j,b,n) to d_out + bf16 copy.
// ---------------------------------------------------------------------------
__global__ __launch_bounds__(256) void softmax_rows(
    const float* __restrict__ score, const unsigned char* __restrict__ mask,
    float* __restrict__ prob, bf16_t* __restrict__ pbf)
{
  const int row = blockIdx.x;            // bn*256 + i
  const int bn = row >> 8, i = row & 255;
  const int b = bn >> 4;
  const int tid = threadIdx.x;
  const float* s = score + (size_t)row * 512;

  float v0 = s[tid] * 0.125f;
  float v1 = s[tid + 256] * 0.125f;
  if (mask[b * 512 + tid])       v0 = -3.0e38f;
  if (mask[b * 512 + tid + 256]) v1 = -3.0e38f;

  __shared__ float sh[256];
  sh[tid] = fmaxf(v0, v1); __syncthreads();
  for (int st = 128; st > 0; st >>= 1) {
    if (tid < st) sh[tid] = fmaxf(sh[tid], sh[tid + st]);
    __syncthreads();
  }
  const float m = sh[0]; __syncthreads();
  const float e0 = __expf(v0 - m), e1 = __expf(v1 - m);
  sh[tid] = e0 + e1; __syncthreads();
  for (int st = 128; st > 0; st >>= 1) {
    if (tid < st) sh[tid] += sh[tid + st];
    __syncthreads();
  }
  const float inv = 1.f / sh[0];
  const float p0 = e0 * inv, p1 = e1 * inv;
  prob[((size_t)(i * 512 + tid)) * 256 + bn]       = p0;
  prob[((size_t)(i * 512 + tid + 256)) * 256 + bn] = p1;
  pbf[(size_t)row * 512 + tid]       = (bf16_t)p0;
  pbf[(size_t)row * 512 + tid + 256] = (bf16_t)p1;
}

// ---------------------------------------------------------------------------
// Elementwise kernels
// ---------------------------------------------------------------------------
__global__ void cvt_bf(const float* __restrict__ s, bf16_t* __restrict__ d, int n) {
  const int i = blockIdx.x * blockDim.x + threadIdx.x;
  if (i < n) d[i] = (bf16_t)s[i];
}

// qa/qb = wq(view) + r_w_bias / r_r_bias  (bf16, (i,b,n,d) view layout)
__global__ void make_qaqb(const bf16_t* __restrict__ wqf,
                          const float* __restrict__ rwb, const float* __restrict__ rrb,
                          bf16_t* __restrict__ qa, bf16_t* __restrict__ qb)
{
  const int t = blockIdx.x * blockDim.x + threadIdx.x;   // 4194304
  const int i = t >> 14, b = (t >> 10) & 15, n = (t >> 6) & 15, d = t & 63;
  const int tq = i * 16 + b;
  const int bp = tq >> 8, qp = tq & 255;
  const float v = (float)wqf[((size_t)((bp * 512 + 256 + qp) * 16 + n)) * 64 + d];
  qa[t] = (bf16_t)(v + rwb[n * 64 + d]);
  qb[t] = (bf16_t)(v + rrb[n * 64 + d]);
}

// r = sigmoid(Sr); z = sigmoid(Sz - bg) stored in-place over Sz; rx = bf16(r*x)
__global__ void gru_rz(const float* __restrict__ Sr, float* __restrict__ Sz,
                       const float* __restrict__ bg, const float* __restrict__ x,
                       bf16_t* __restrict__ rxb, int n)
{
  const int t = blockIdx.x * blockDim.x + threadIdx.x;
  if (t >= n) return;
  const float r = 1.f / (1.f + __expf(-Sr[t]));
  const float z = 1.f / (1.f + __expf(-(Sz[t] - bg[t & 1023])));
  Sz[t] = z;
  rxb[t] = (bf16_t)(r * x[t]);
}

// out = (1-z)*x + z*tanh(Sg)
__global__ void gru_fin(const float* __restrict__ Sg, const float* __restrict__ Z,
                        const float* __restrict__ x, float* __restrict__ out,
                        bf16_t* __restrict__ outb, int n)
{
  const int t = blockIdx.x * blockDim.x + threadIdx.x;
  if (t >= n) return;
  const float z = Z[t];
  const float o = (1.f - z) * x[t] + z * tanhf(Sg[t]);
  out[t] = o;
  if (outb) outb[t] = (bf16_t)o;
}

// fwd = relu(S + bias) -> bf16
__global__ void fc_post(const float* __restrict__ S, const float* __restrict__ bias,
                        bf16_t* __restrict__ fb, int n)
{
  const int t = blockIdx.x * blockDim.x + threadIdx.x;
  if (t >= n) return;
  float v = S[t] + bias[t & 1023];
  fb[t] = (bf16_t)(v > 0.f ? v : 0.f);
}

// ---------------------------------------------------------------------------
// Host orchestration
// ---------------------------------------------------------------------------
extern "C" void kernel_launch(void* const* d_in, const int* in_sizes, int n_in,
                              void* d_out, int out_size, void* d_ws, size_t ws_size,
                              hipStream_t stream)
{
  (void)in_sizes; (void)n_in; (void)out_size; (void)ws_size;
  const float* value = (const float*)d_in[0];
  const float* query = (const float*)d_in[1];
  const unsigned char* mask = (const unsigned char*)d_in[2];   // jnp bool, 1B
  const float* rmat = (const float*)d_in[3];
  const float* rwb  = (const float*)d_in[4];
  const float* rrb  = (const float*)d_in[5];
  const float* Wq   = (const float*)d_in[6];
  const float* Wk   = (const float*)d_in[7];
  const float* Wv   = (const float*)d_in[8];
  const float* Wrn  = (const float*)d_in[9];
  const float* ln1g = (const float*)d_in[10];
  const float* ln1b = (const float*)d_in[11];
  const float* ln2g = (const float*)d_in[12];
  const float* ln2b = (const float*)d_in[13];
  const float* lnkg = (const float*)d_in[14];
  const float* lnkb = (const float*)d_in[15];
  const float* G1[6]; for (int i = 0; i < 6; ++i) G1[i] = (const float*)d_in[16 + i];
  const float* g1bg = (const float*)d_in[22];
  const float* G2[6]; for (int i = 0; i < 6; ++i) G2[i] = (const float*)d_in[23 + i];
  const float* g2bg = (const float*)d_in[29];
  const float* fcW  = (const float*)d_in[30];
  const float* fcb  = (const float*)d_in[31];

  float* outp  = (float*)d_out;              // (B,QLEN,D) = 4194304 f32
  float* probp = outp + 4194304;             // (256,512,16,16) = 33554432 f32

  // ---- workspace carve ----
  char* w = (char*)d_ws;
  size_t off = 0;
  auto alloc = [&](size_t bytes) -> void* {
    void* p = w + off;
    off = (off + bytes + 255) & ~(size_t)255;
    return p;
  };
  bf16_t* kvb  = (bf16_t*)alloc(8388608ull * 2);    // (B,KLEN,D) bf16
  bf16_t* qryb = (bf16_t*)alloc(4194304ull * 2);    // query bf16
  bf16_t* rb   = (bf16_t*)alloc(524288ull * 2);     // r bf16
  bf16_t* wqw  = (bf16_t*)alloc(4096ull * 2);
  bf16_t* wkw  = (bf16_t*)alloc(4096ull * 2);
  bf16_t* wvw  = (bf16_t*)alloc(4096ull * 2);
  bf16_t* wrnb = (bf16_t*)alloc(1048576ull * 2);
  bf16_t* gwb[13];
  for (int i = 0; i < 13; ++i) gwb[i] = (bf16_t*)alloc(1048576ull * 2); // g1[6],g2[6],fc
  bf16_t* wkb  = (bf16_t*)alloc(8388608ull * 2);    // k proj (b,k,n,e)
  bf16_t* wvT  = (bf16_t*)alloc(8388608ull * 2);    // v proj transposed per (b,n)
  bf16_t* wqf  = (bf16_t*)alloc(8388608ull * 2);    // q proj full kv rows
  bf16_t* qa   = (bf16_t*)alloc(4194304ull * 2);    // wq+rwb (view layout)
  bf16_t* qb   = (bf16_t*)alloc(4194304ull * 2);    // wq+rrb
  bf16_t* rkb  = (bf16_t*)alloc(524288ull * 2);     // rk bf16 (512,1024)
  bf16_t* pbf  = (bf16_t*)alloc(33554432ull * 2);   // prob bf16
  bf16_t* vecb = (bf16_t*)alloc(4194304ull * 2);    // attn out bf16
  float*  score = (float*)alloc(33554432ull * 4);   // 134 MB, dead after softmax

  // GRU-phase temporaries alias the (dead) score region:
  char* sreg = (char*)score;
  float*  Sr   = (float*)(sreg);
  float*  Sz   = (float*)(sreg + 1ull * 16777216);
  float*  Sg   = (float*)(sreg + 2ull * 16777216);
  float*  hbuf = (float*)(sreg + 3ull * 16777216);
  bf16_t* rxb  = (bf16_t*)(sreg + 4ull * 16777216);
  bf16_t* hb   = (bf16_t*)(sreg + 4ull * 16777216 + 1ull * 8388608);
  bf16_t* hlnb = (bf16_t*)(sreg + 4ull * 16777216 + 2ull * 8388608);
  bf16_t* fwdb = (bf16_t*)(sreg + 4ull * 16777216 + 3ull * 8388608);

  const int T = 256;
  // ---- 1. conversions to bf16 ----
  cvt_bf<<<16384, T, 0, stream>>>(query, qryb, 4194304);
  cvt_bf<<<2048,  T, 0, stream>>>(rmat, rb, 524288);
  cvt_bf<<<16,    T, 0, stream>>>(Wq, wqw, 4096);
  cvt_bf<<<16,    T, 0, stream>>>(Wk, wkw, 4096);
  cvt_bf<<<16,    T, 0, stream>>>(Wv, wvw, 4096);
  cvt_bf<<<4096,  T, 0, stream>>>(Wrn, wrnb, 1048576);
  for (int i = 0; i < 6; ++i) cvt_bf<<<4096, T, 0, stream>>>(G1[i], gwb[i], 1048576);
  for (int i = 0; i < 6; ++i) cvt_bf<<<4096, T, 0, stream>>>(G2[i], gwb[6 + i], 1048576);
  cvt_bf<<<4096, T, 0, stream>>>(fcW, gwb[12], 1048576);

  // ---- 2. layernorms into kv = concat(LN(value), LN(query)) ----
  ln_rows<<<4096, T, 0, stream>>>(value, lnkg, lnkb, kvb, 256, 512, 0);
  ln_rows<<<4096, T, 0, stream>>>(query, ln1g, ln1b, kvb, 256, 512, 256);

  // ---- 3. per-head projections (131072 rows of 64) ----
  proj16<<<1024, T, 0, stream>>>(kvb, wkw, wkb, 0, 131072);
  proj16<<<1024, T, 0, stream>>>(kvb, wvw, wvT, 1, 131072);
  proj16<<<1024, T, 0, stream>>>(kvb, wqw, wqf, 0, 131072);

  // ---- 4. rk = r @ Wr_net^T -> bf16 (512x1024x1024) ----
  gemm64x64<<<16, T, 0, stream>>>(rb, 1024, wrnb, 1024, nullptr, rkb, 1024,
                                  512, 1024, 1024, 0);

  // ---- 5. qa/qb = wq(view) + biases ----
  make_qaqb<<<16384, T, 0, stream>>>(wqf, rwb, rrb, qa, qb);

  // ---- 6. attention scores: AC then rel-shifted BD scatter-add ----
  attn_ac<<<4096, T, 0, stream>>>(qa, wkb, score);
  attn_bd<<<4096, T, 0, stream>>>(qb, rkb, score);

  // ---- 7. masked softmax -> prob (f32 to d_out) + bf16 copy ----
  softmax_rows<<<65536, T, 0, stream>>>(score, mask, probp, pbf);

  // ---- 8. vec = prob @ wv ----
  attn_vec<<<512, T, 0, stream>>>(pbf, wvT, vecb);

  // ---- 9. GRU1 (x=query, y=attn) ----
  const int GB = 128;  // (4096/64)*(1024/64)=1024 waves, 8/block
  gemm64x64<<<GB, T, 0, stream>>>(vecb, 1024, gwb[0], 1024, Sr, nullptr, 1024, 4096, 1024, 1024, 0);
  gemm64x64<<<GB, T, 0, stream>>>(qryb, 1024, gwb[1], 1024, Sr, nullptr, 1024, 4096, 1024, 1024, 1);
  gemm64x64<<<GB, T, 0, stream>>>(vecb, 1024, gwb[2], 1024, Sz, nullptr, 1024, 4096, 1024, 1024, 0);
  gemm64x64<<<GB, T, 0, stream>>>(qryb, 1024, gwb[3], 1024, Sz, nullptr, 1024, 4096, 1024, 1024, 1);
  gemm64x64<<<GB, T, 0, stream>>>(vecb, 1024, gwb[4], 1024, Sg, nullptr, 1024, 4096, 1024, 1024, 0);
  gru_rz<<<16384, T, 0, stream>>>(Sr, Sz, g1bg, query, rxb, 4194304);
  gemm64x64<<<GB, T, 0, stream>>>(rxb, 1024, gwb[5], 1024, Sg, nullptr, 1024, 4096, 1024, 1024, 1);
  gru_fin<<<16384, T, 0, stream>>>(Sg, Sz, query, hbuf, hb, 4194304);

  // ---- 10. LN2 + FC ----
  ln_rows<<<4096, T, 0, stream>>>(hbuf, ln2g, ln2b, hlnb, 4096, 0, 0);
  gemm64x64<<<GB, T, 0, stream>>>(hlnb, 1024, gwb[12], 1024, Sr, nullptr, 1024, 4096, 1024, 1024, 0);
  fc_post<<<16384, T, 0, stream>>>(Sr, fcb, fwdb, 4194304);

  // ---- 11. GRU2 (x=h, y=fwd) -> out ----
  gemm64x64<<<GB, T, 0, stream>>>(fwdb, 1024, gwb[6],  1024, Sr, nullptr, 1024, 4096, 1024, 1024, 0);
  gemm64x64<<<GB, T, 0, stream>>>(hb,   1024, gwb[7],  1024, Sr, nullptr, 1024, 4096, 1024, 1024, 1);
  gemm64x64<<<GB, T, 0, stream>>>(fwdb, 1024, gwb[8],  1024, Sz, nullptr, 1024, 4096, 1024, 1024, 0);
  gemm64x64<<<GB, T, 0, stream>>>(hb,   1024, gwb[9],  1024, Sz, nullptr, 1024, 4096, 1024, 1024, 1);
  gemm64x64<<<GB, T, 0, stream>>>(fwdb, 1024, gwb[10], 1024, Sg, nullptr, 1024, 4096, 1024, 1024, 0);
  gru_rz<<<16384, T, 0, stream>>>(Sr, Sz, g2bg, hbuf, rxb, 4194304);
  gemm64x64<<<GB, T, 0, stream>>>(rxb, 1024, gwb[11], 1024, Sg, nullptr, 1024, 4096, 1024, 1024, 1);
  gru_fin<<<16384, T, 0, stream>>>(Sg, Sz, hbuf, outp, nullptr, 4194304);
}
